// MultiHeadAttention_22179211117104
// MI455X (gfx1250) — matrix operand order repaired and checked by reading, not compiled, not run on hardware
//
#include <hip/hip_runtime.h>

#define D_EMBED 768
#define N_QKV   2304
#define HEADS   12
#define HD      64
#define SEQ_T   2048
#define BATCH   4
#define M_TOT   (BATCH * SEQ_T)

typedef __attribute__((ext_vector_type(16))) __bf16 v16bf;
typedef __attribute__((ext_vector_type(8)))  float  v8f;

union BF16Frag {
    v16bf          v;
    unsigned short u[16];
    uint4          q[2];
};

__device__ __forceinline__ unsigned short f32_to_bf16(float f) {
    union { float f; unsigned int u; } c;
    c.f = f;
    unsigned int x = c.u;
    x += 0x7FFFu + ((x >> 16) & 1u);   // round-to-nearest-even
    return (unsigned short)(x >> 16);
}

#if __has_builtin(__builtin_amdgcn_cvt_pk_bf16_f32)
typedef __attribute__((ext_vector_type(2))) __bf16 bf16x2;
__device__ __forceinline__ unsigned int pack_bf16x2(float lo, float hi) {
    union { bf16x2 v; unsigned int u; } c;
    c.v = __builtin_amdgcn_cvt_pk_bf16_f32(lo, hi);
    return c.u;
}
#else
__device__ __forceinline__ unsigned int pack_bf16x2(float lo, float hi) {
    return (unsigned int)f32_to_bf16(lo) | ((unsigned int)f32_to_bf16(hi) << 16);
}
#endif

// 8x4 u16 register transpose via v_perm_b32 (used by attention V^T staging).
__device__ __forceinline__ void transpose_8x4_u16(const uint2 r[8], uint4 out[4]) {
#pragma unroll
    for (int p = 0; p < 4; ++p) {
        unsigned a0 = r[2 * p].x,     a1 = r[2 * p].y;
        unsigned b0 = r[2 * p + 1].x, b1 = r[2 * p + 1].y;
        (&out[0].x)[p] = __builtin_amdgcn_perm(b0, a0, 0x05040100u);
        (&out[1].x)[p] = __builtin_amdgcn_perm(b0, a0, 0x07060302u);
        (&out[2].x)[p] = __builtin_amdgcn_perm(b1, a1, 0x05040100u);
        (&out[3].x)[p] = __builtin_amdgcn_perm(b1, a1, 0x07060302u);
    }
}

// ---------------------------------------------------------------------------
// Kernel 0a: bulk fp32 -> bf16 conversion (row-major preserved), for x.
// ---------------------------------------------------------------------------
__global__ __launch_bounds__(256) void cvt_bf16_kernel(
    const float* __restrict__ in, unsigned short* __restrict__ out)
{
    size_t idx = ((size_t)blockIdx.x * 256 + threadIdx.x) * 8;
    float4 a = reinterpret_cast<const float4*>(in + idx)[0];
    float4 b = reinterpret_cast<const float4*>(in + idx)[1];
    uint4 o;
    o.x = pack_bf16x2(a.x, a.y);
    o.y = pack_bf16x2(a.z, a.w);
    o.z = pack_bf16x2(b.x, b.y);
    o.w = pack_bf16x2(b.z, b.w);
    *reinterpret_cast<uint4*>(out + idx) = o;
}

// ---------------------------------------------------------------------------
// Kernel 0b: fp32 [K][N] -> bf16 [N][K] tiled transpose+convert, for weights.
// 32x32 tiles through LDS; coalesced on both sides. grid = (N/32, K/32).
// ---------------------------------------------------------------------------
__global__ __launch_bounds__(256) void transpose_cvt_kernel(
    const float* __restrict__ in, unsigned short* __restrict__ out,
    int K, int N)
{
    __shared__ unsigned short t[32][36];   // row stride 72B (8B-aligned chunks)
    const int tid = threadIdx.x;
    const int n0 = blockIdx.x * 32, k0 = blockIdx.y * 32;
    {
        int ki = tid >> 3;            // 0..31
        int nj = (tid & 7) * 4;       // 4 consecutive n
        float4 f = *reinterpret_cast<const float4*>(in + (size_t)(k0 + ki) * N + n0 + nj);
        t[nj + 0][ki] = f32_to_bf16(f.x);
        t[nj + 1][ki] = f32_to_bf16(f.y);
        t[nj + 2][ki] = f32_to_bf16(f.z);
        t[nj + 3][ki] = f32_to_bf16(f.w);
    }
    __syncthreads();
    {
        int ni = tid >> 3;            // 0..31
        int kj = (tid & 7) * 4;       // 4 consecutive k
        *reinterpret_cast<uint2*>(out + (size_t)(n0 + ni) * K + k0 + kj) =
            *reinterpret_cast<const uint2*>(&t[ni][kj]);
    }
}

// ---------------------------------------------------------------------------
// Kernel 1: QKV projection. C[8192 x 2304] = xb @ wqT^T + b  (bf16 inputs)
// Weights pre-transposed [n][k] in global: NO LDS, NO barriers — both A and B
// fragments are contiguous global b128 loads (served by WGP$/L2).
// Block tile 128x256, 8 waves (2 x 4), per-wave 64x64 via 4x4 WMMA frags.
// ---------------------------------------------------------------------------
__global__ __launch_bounds__(256) void qkv_proj_kernel(
    const unsigned short* __restrict__ xb, const unsigned short* __restrict__ wqT,
    const float* __restrict__ bias,
    unsigned short* __restrict__ qb, unsigned short* __restrict__ kb,
    unsigned short* __restrict__ vb)
{
    constexpr int BK = 32, BN = 256;
    const int tid  = threadIdx.x;
    const int lane = tid & 31;
    const int wave = tid >> 5;
    const int wm   = wave >> 2;        // 0..1  -> M offset wm*64
    const int wn   = wave & 3;         // 0..3  -> N offset wn*64
    const int n0   = blockIdx.x * BN;
    const int m0   = blockIdx.y * 128;
    const int lrow = lane & 15;
    const int hi   = lane >> 4;
    const int akb  = hi ? 8 : 0;
    const int bkb  = hi ? 8 : 0;

    v8f acc[4][4];
#pragma unroll
    for (int i = 0; i < 4; ++i)
#pragma unroll
        for (int j = 0; j < 4; ++j) acc[i][j] = (v8f)0.f;

    const unsigned short* arow0 =
        xb + (size_t)(m0 + wm * 64 + lrow) * D_EMBED + akb;
    const unsigned short* brow0 =
        wqT + (size_t)(n0 + wn * 64 + lrow) * D_EMBED + bkb;

    for (int k0 = 0; k0 < D_EMBED; k0 += BK) {
        if (k0 + BK < D_EMBED)
            __builtin_prefetch(arow0 + k0 + BK, 0, 1);
        BF16Frag bf[4];
#pragma unroll
        for (int nf = 0; nf < 4; ++nf) {
            const unsigned short* p = brow0 + (size_t)(nf * 16) * D_EMBED + k0;
            bf[nf].q[0] = *reinterpret_cast<const uint4*>(p);
            bf[nf].q[1] = *reinterpret_cast<const uint4*>(p + 16);
        }
#pragma unroll
        for (int mf = 0; mf < 4; ++mf) {
            const unsigned short* p = arow0 + (size_t)(mf * 16) * D_EMBED + k0;
            BF16Frag af;
            af.q[0] = *reinterpret_cast<const uint4*>(p);
            af.q[1] = *reinterpret_cast<const uint4*>(p + 16);
#pragma unroll
            for (int nf = 0; nf < 4; ++nf)
                acc[mf][nf] = __builtin_amdgcn_wmma_f32_16x16x32_bf16(
                    false, af.v, false, bf[nf].v, (short)0, acc[mf][nf], false, false);
        }
    }

    // --- epilogue: bias + scatter to Q/K/V [B,H,T,hd] as bf16 ---
    const int seg = n0 / D_EMBED;      // 256 | 768 => uniform per block
    unsigned short* dst = (seg == 0) ? qb : ((seg == 1) ? kb : vb);
#pragma unroll
    for (int mf = 0; mf < 4; ++mf)
#pragma unroll
        for (int nf = 0; nf < 4; ++nf)
#pragma unroll
            for (int r = 0; r < 8; ++r) {
                int row = m0 + wm * 64 + mf * 16 + r + hi * 8;
                int col = n0 + wn * 64 + nf * 16 + lrow;
                float val = acc[mf][nf][r] + bias[col];
                int cw = col - seg * D_EMBED;
                int h = cw >> 6, dpos = cw & 63;
                int b = row >> 11, t = row & (SEQ_T - 1);
                size_t idx = (((size_t)b * HEADS + h) * SEQ_T + t) * HD + dpos;
                dst[idx] = f32_to_bf16(val);
            }
}

// ---------------------------------------------------------------------------
// Kernel 2: flash attention per (b, h, 64-query tile). 4 waves x 16 rows.
// ---------------------------------------------------------------------------
__global__ __launch_bounds__(128) void attention_kernel(
    const unsigned short* __restrict__ qb, const unsigned short* __restrict__ kb,
    const unsigned short* __restrict__ vb, unsigned short* __restrict__ attn)
{
    constexpr int VSTR = 80, PSTR = 80;            // *2 bytes = 160, 16B aligned
    __shared__ unsigned short Vt[64 * VSTR];       // V^T tile: [hd][key]
    __shared__ unsigned short Ps[4][16 * PSTR];    // per-wave P: [row][key]

    const int tid  = threadIdx.x;
    const int lane = tid & 31;
    const int wave = tid >> 5;
    const int lrow = lane & 15;
    const int hi   = lane >> 4;
    const int akb  = hi ? 8 : 0;
    const int bkb  = hi ? 8 : 0;
    const int h  = blockIdx.y, b = blockIdx.z;
    const int q0 = blockIdx.x * 64;
    const size_t head_base = ((size_t)b * HEADS + h) * SEQ_T * HD;
    const unsigned short* Q = qb + head_base;
    const unsigned short* K = kb + head_base;
    const unsigned short* V = vb + head_base;

    BF16Frag qa[2];
    {
        const unsigned short* qp = Q + (size_t)(q0 + wave * 16 + lrow) * HD;
#pragma unroll
        for (int kk = 0; kk < 2; ++kk) {
            const unsigned short* p = qp + kk * 32 + akb;
            qa[kk].q[0] = *reinterpret_cast<const uint4*>(p);
            qa[kk].q[1] = *reinterpret_cast<const uint4*>(p + 16);
        }
    }

    v8f o[4];
#pragma unroll
    for (int of = 0; of < 4; ++of) o[of] = (v8f)0.f;
    float mrow[8], lsum[8];
#pragma unroll
    for (int r = 0; r < 8; ++r) { mrow[r] = -1e30f; lsum[r] = 0.f; }
    const float SSCALE = 0.125f * 1.44269504088896341f;  // 1/sqrt(64) * log2(e)

    const int keybase = (tid & 7) * 8;   // 8 consecutive keys per thread
    const int hdbase  = (tid >> 3) * 4;  // 4 hd values per thread

    for (int kt = 0; kt < SEQ_T; kt += 64) {
        __syncthreads();
        // --- stage V^T tile: 8x(b64) -> v_perm transpose -> 4x(ds_store_b128)
        {
            const unsigned short* vp = V + (size_t)(kt + keybase) * HD + hdbase;
            uint2 r[8];
#pragma unroll
            for (int kk = 0; kk < 8; ++kk)
                r[kk] = *reinterpret_cast<const uint2*>(vp + (size_t)kk * HD);
            uint4 col[4];
            transpose_8x4_u16(r, col);
#pragma unroll
            for (int j = 0; j < 4; ++j)
                *reinterpret_cast<uint4*>(&Vt[(hdbase + j) * VSTR + keybase]) = col[j];
        }
        __syncthreads();

        // --- scores: 16 q-rows x 64 keys, K-frags direct from global ---
        v8f s[4];
#pragma unroll
        for (int sf = 0; sf < 4; ++sf) {
            s[sf] = (v8f)0.f;
#pragma unroll
            for (int kk = 0; kk < 2; ++kk) {
                BF16Frag kf;
                const unsigned short* p =
                    K + (size_t)(kt + sf * 16 + lrow) * HD + kk * 32 + bkb;
                kf.q[0] = reinterpret_cast<const uint4*>(p)[0];
                kf.q[1] = *reinterpret_cast<const uint4*>(p + 16);
                s[sf] = __builtin_amdgcn_wmma_f32_16x16x32_bf16(
                    false, qa[kk].v, false, kf.v, (short)0, s[sf], false, false);
            }
        }

        // --- online softmax over this 64-key chunk (exp2 domain) ---
        float mnew[8], alpha[8], rsum[8];
#pragma unroll
        for (int r = 0; r < 8; ++r) {
            float v0 = fmaxf(fmaxf(s[0][r], s[1][r]), fmaxf(s[2][r], s[3][r]));
#pragma unroll
            for (int off = 1; off < 16; off <<= 1)
                v0 = fmaxf(v0, __shfl_xor(v0, off, 32));
            v0 *= SSCALE;
            mnew[r]  = fmaxf(mrow[r], v0);
            alpha[r] = exp2f(mrow[r] - mnew[r]);
            mrow[r]  = mnew[r];
            rsum[r]  = 0.f;
        }
#pragma unroll
        for (int sf = 0; sf < 4; ++sf)
#pragma unroll
            for (int r = 0; r < 8; ++r) {
                float p = exp2f(s[sf][r] * SSCALE - mnew[r]);
                rsum[r] += p;
                Ps[wave][(r + hi * 8) * PSTR + sf * 16 + lrow] = f32_to_bf16(p);
            }
#pragma unroll
        for (int r = 0; r < 8; ++r) {
            float v0 = rsum[r];
#pragma unroll
            for (int off = 1; off < 16; off <<= 1) v0 += __shfl_xor(v0, off, 32);
            lsum[r] = lsum[r] * alpha[r] + v0;
#pragma unroll
            for (int of = 0; of < 4; ++of) o[of][r] *= alpha[r];
        }

        // --- O += P @ V  (K = 64 keys -> two 16x16x32 steps) ---
#pragma unroll
        for (int kk = 0; kk < 2; ++kk) {
            BF16Frag pa;
            const unsigned short* pp = &Ps[wave][lrow * PSTR + kk * 32 + akb];
            pa.q[0] = *reinterpret_cast<const uint4*>(pp);
            pa.q[1] = *reinterpret_cast<const uint4*>(pp + 16);
#pragma unroll
            for (int of = 0; of < 4; ++of) {
                BF16Frag vf;
                const unsigned short* vp2 = &Vt[(of * 16 + lrow) * VSTR + kk * 32 + bkb];
                vf.q[0] = reinterpret_cast<const uint4*>(vp2)[0];
                vf.q[1] = *reinterpret_cast<const uint4*>(vp2 + 16);
                o[of] = __builtin_amdgcn_wmma_f32_16x16x32_bf16(
                    false, pa.v, false, vf.v, (short)0, o[of], false, false);
            }
        }
    }

    // --- normalize + write attn [B, T, D] bf16 ---
#pragma unroll
    for (int r = 0; r < 8; ++r) {
        float inv  = 1.0f / lsum[r];
        int qrow   = q0 + wave * 16 + r + hi * 8;
        size_t base = ((size_t)b * SEQ_T + qrow) * D_EMBED + h * HD;
#pragma unroll
        for (int of = 0; of < 4; ++of)
            attn[base + of * 16 + lrow] = f32_to_bf16(o[of][r] * inv);
    }
}

// ---------------------------------------------------------------------------
// Kernel 3: output projection. out[8192 x 768] = attn @ woT^T + b_out (fp32)
// Same LDS-free structure as kernel 1.
// ---------------------------------------------------------------------------
__global__ __launch_bounds__(256) void out_proj_kernel(
    const unsigned short* __restrict__ attn, const unsigned short* __restrict__ woT,
    const float* __restrict__ bias, float* __restrict__ out)
{
    constexpr int BK = 32, BN = 256;
    const int tid  = threadIdx.x;
    const int lane = tid & 31;
    const int wave = tid >> 5;
    const int wm   = wave >> 2;
    const int wn   = wave & 3;
    const int n0   = blockIdx.x * BN;
    const int m0   = blockIdx.y * 128;
    const int lrow = lane & 15;
    const int hi   = lane >> 4;
    const int akb  = hi ? 8 : 0;
    const int bkb  = hi ? 8 : 0;

    v8f acc[4][4];
#pragma unroll
    for (int i = 0; i < 4; ++i)
#pragma unroll
        for (int j = 0; j < 4; ++j) acc[i][j] = (v8f)0.f;

    const unsigned short* arow0 =
        attn + (size_t)(m0 + wm * 64 + lrow) * D_EMBED + akb;
    const unsigned short* brow0 =
        woT + (size_t)(n0 + wn * 64 + lrow) * D_EMBED + bkb;

    for (int k0 = 0; k0 < D_EMBED; k0 += BK) {
        if (k0 + BK < D_EMBED)
            __builtin_prefetch(arow0 + k0 + BK, 0, 1);
        BF16Frag bf[4];
#pragma unroll
        for (int nf = 0; nf < 4; ++nf) {
            const unsigned short* p = brow0 + (size_t)(nf * 16) * D_EMBED + k0;
            bf[nf].q[0] = *reinterpret_cast<const uint4*>(p);
            bf[nf].q[1] = *reinterpret_cast<const uint4*>(p + 16);
        }
#pragma unroll
        for (int mf = 0; mf < 4; ++mf) {
            const unsigned short* p = arow0 + (size_t)(mf * 16) * D_EMBED + k0;
            BF16Frag af;
            af.q[0] = *reinterpret_cast<const uint4*>(p);
            af.q[1] = *reinterpret_cast<const uint4*>(p + 16);
#pragma unroll
            for (int nf = 0; nf < 4; ++nf)
                acc[mf][nf] = __builtin_amdgcn_wmma_f32_16x16x32_bf16(
                    false, af.v, false, bf[nf].v, (short)0, acc[mf][nf], false, false);
        }
    }

#pragma unroll
    for (int mf = 0; mf < 4; ++mf)
#pragma unroll
        for (int nf = 0; nf < 4; ++nf)
#pragma unroll
            for (int r = 0; r < 8; ++r) {
                int row = m0 + wm * 64 + mf * 16 + r + hi * 8;
                int col = n0 + wn * 64 + nf * 16 + lrow;
                out[(size_t)row * D_EMBED + col] = acc[mf][nf][r] + bias[col];
            }
}

// ---------------------------------------------------------------------------
extern "C" void kernel_launch(void* const* d_in, const int* in_sizes, int n_in,
                              void* d_out, int out_size, void* d_ws, size_t ws_size,
                              hipStream_t stream)
{
    const float* x     = (const float*)d_in[0];
    const float* w_qkv = (const float*)d_in[1];
    const float* b_qkv = (const float*)d_in[2];
    const float* w_out = (const float*)d_in[3];
    const float* b_out = (const float*)d_in[4];
    float* out = (float*)d_out;

    const size_t nQKV = (size_t)BATCH * HEADS * SEQ_T * HD;   // 6,291,456
    const size_t nX   = (size_t)M_TOT * D_EMBED;              // 6,291,456
    const size_t nWQ  = (size_t)D_EMBED * N_QKV;              // 1,769,472
    const size_t nWO  = (size_t)D_EMBED * D_EMBED;            //   589,824

    unsigned short* qb  = (unsigned short*)d_ws;
    unsigned short* kbp = qb  + nQKV;
    unsigned short* vbp = kbp + nQKV;
    unsigned short* xa  = vbp + nQKV;     // x_bf16; reused as attn after QKV GEMM
    unsigned short* wqT = xa  + nX;       // w_qkv transposed [2304][768] bf16
    unsigned short* woT = wqT + nWQ;      // w_out transposed [768][768] bf16

    cvt_bf16_kernel<<<dim3((unsigned)(nX / 2048)), 256, 0, stream>>>(x, xa);
    transpose_cvt_kernel<<<dim3(N_QKV / 32, D_EMBED / 32), 256, 0, stream>>>(
        w_qkv, wqT, D_EMBED, N_QKV);
    transpose_cvt_kernel<<<dim3(D_EMBED / 32, D_EMBED / 32), 256, 0, stream>>>(
        w_out, woT, D_EMBED, D_EMBED);

    qkv_proj_kernel<<<dim3(N_QKV / 256, M_TOT / 128), 256, 0, stream>>>(
        xa, wqT, b_qkv, qb, kbp, vbp);
    attention_kernel<<<dim3(SEQ_T / 64, HEADS, BATCH), 128, 0, stream>>>(
        qb, kbp, vbp, xa /* attn reuses x_bf16 */);
    out_proj_kernel<<<dim3(D_EMBED / 256, M_TOT / 128), 256, 0, stream>>>(
        xa, woT, b_out, out);
}
